// ZeroInflatedGAT_50337016709816
// MI455X (gfx1250) — matrix operand order repaired and checked by reading, not compiled, not run on hardware
//
#include <hip/hip_runtime.h>
#include <math.h>

typedef _Float16 half16 __attribute__((ext_vector_type(16)));
typedef float    float8 __attribute__((ext_vector_type(8)));

#define NEG_SLOPE 0.2f

// ---------------------------------------------------------------------------
// GEMM: H[n x 32] = A[n x K] @ W[K x 32]   (K in {32, 64})
// One wave per 16-row tile; two 16x16 WMMA accumulators cover the 32 columns.
// f16 inputs, f32 accumulation via v_wmma_f32_16x16x32_f16.
// W is staged in LDS once per block (8 waves share it).
// ---------------------------------------------------------------------------
__global__ __launch_bounds__(256) void gat_gemm_wmma(
    const float* __restrict__ A, const float* __restrict__ W,
    float* __restrict__ H, int n, int K) {
  __shared__ float Wlds[64 * 32];          // up to 8 KB

  // cooperative W load (K*32 floats = K*8 float4)
  int nv4 = K * 8;
  for (int i = threadIdx.x; i < nv4; i += 256)
    ((float4*)Wlds)[i] = ((const float4*)W)[i];
  __syncthreads();

  int gid  = blockIdx.x * blockDim.x + threadIdx.x;
  int tile = gid >> 5;                     // one 16-row tile per wave
  int lane = threadIdx.x & 31;
  int hi   = lane >> 4;                    // lane half (0 or 1)
  int l15  = lane & 15;

  if (tile * 16 >= n) return;              // after the barrier: safe

  int row = tile * 16 + l15;
  if (row >= n) row = n - 1;               // tail clamp (stores are guarded)

  float8 acc0 = {};
  float8 acc1 = {};
  const float* arow = A + (size_t)row * K;

  for (int k0 = 0; k0 < K; k0 += 32) {
    // A fragment: 16-bit A 16x32 layout (ISA 7.12.2)
    int ka = k0 + hi * 8;
    float4 a0 = *(const float4*)(arow + ka);
    float4 a1 = *(const float4*)(arow + ka + 4);
    float4 a2 = *(const float4*)(arow + ka + 16);
    float4 a3 = *(const float4*)(arow + ka + 20);
    half16 a;
    a[0]  = (_Float16)a0.x; a[1]  = (_Float16)a0.y;
    a[2]  = (_Float16)a0.z; a[3]  = (_Float16)a0.w;
    a[4]  = (_Float16)a1.x; a[5]  = (_Float16)a1.y;
    a[6]  = (_Float16)a1.z; a[7]  = (_Float16)a1.w;
    a[8]  = (_Float16)a2.x; a[9]  = (_Float16)a2.y;
    a[10] = (_Float16)a2.z; a[11] = (_Float16)a2.w;
    a[12] = (_Float16)a3.x; a[13] = (_Float16)a3.y;
    a[14] = (_Float16)a3.z; a[15] = (_Float16)a3.w;

    // B fragments from LDS: lane<16 -> K k0..k0+7,k0+16..k0+23, lane>=16 -> K k0+8..k0+15,k0+24..k0+31
    int kb = k0 + hi * 8;
    half16 b0, b1;
#pragma unroll
    for (int i = 0; i < 16; ++i) {
      const float* wr = Wlds + (kb + i + (i & 8)) * 32;
      b0[i] = (_Float16)wr[l15];
      b1[i] = (_Float16)wr[l15 + 16];
    }

    acc0 = __builtin_amdgcn_wmma_f32_16x16x32_f16(false, a, false, b0,
                                                  (short)0, acc0, false, false);
    acc1 = __builtin_amdgcn_wmma_f32_16x16x32_f16(false, a, false, b1,
                                                  (short)0, acc1, false, false);
  }

  // C/D layout: lane holds col l15; VGPR r holds row r + 8*hi
  if (tile * 16 + 16 <= n) {               // full tile: unguarded stores
#pragma unroll
    for (int r = 0; r < 8; ++r) {
      int m = tile * 16 + r + hi * 8;
      H[(size_t)m * 32 + l15]      = acc0[r];
      H[(size_t)m * 32 + l15 + 16] = acc1[r];
    }
  } else {
#pragma unroll
    for (int r = 0; r < 8; ++r) {
      int m = tile * 16 + r + hi * 8;
      if (m < n) {
        H[(size_t)m * 32 + l15]      = acc0[r];
        H[(size_t)m * 32 + l15 + 16] = acc1[r];
      }
    }
  }
}

// ---------------------------------------------------------------------------
// Edge-list prep: int64 -> int32, self-loops materialized. Run once.
// ---------------------------------------------------------------------------
__global__ void edge_prep(const long long* __restrict__ src,
                          const long long* __restrict__ dst, int E, int n,
                          int* __restrict__ s32, int* __restrict__ d32) {
  int e = blockIdx.x * blockDim.x + threadIdx.x;
  if (e >= E + n) return;
  if (e < E) {
    s32[e] = (int)src[e];
    d32[e] = (int)dst[e];
  } else {
    s32[e] = e - E;
    d32[e] = e - E;
  }
}

// ---------------------------------------------------------------------------
// Per-node attention prep + init: e_src/e_dst dot products, amax=-inf,
// denom=0, accum row = 0.
// ---------------------------------------------------------------------------
__global__ void att_init(const float* __restrict__ H,
                         const float* __restrict__ a_s,
                         const float* __restrict__ a_d,
                         float* __restrict__ e_src, float* __restrict__ e_dst,
                         float* __restrict__ amax, float* __restrict__ denom,
                         float* __restrict__ accum, int n) {
  int i = blockIdx.x * blockDim.x + threadIdx.x;
  if (i >= n) return;
  const float4* hr = (const float4*)(H + (size_t)i * 32);
  const float4* as4 = (const float4*)a_s;
  const float4* ad4 = (const float4*)a_d;
  float es = 0.f, ed = 0.f;
#pragma unroll
  for (int j = 0; j < 8; ++j) {
    float4 v = hr[j], s = as4[j], d = ad4[j];
    es += v.x * s.x + v.y * s.y + v.z * s.z + v.w * s.w;
    ed += v.x * d.x + v.y * d.y + v.z * d.z + v.w * d.w;
  }
  e_src[i] = es;
  e_dst[i] = ed;
  amax[i]  = -__builtin_inff();
  denom[i] = 0.f;
  float4* ar = (float4*)(accum + (size_t)i * 32);
  float4 z = {0.f, 0.f, 0.f, 0.f};
#pragma unroll
  for (int j = 0; j < 8; ++j) ar[j] = z;
}

// float atomic max via signed/unsigned int atomics (init value is -inf)
__device__ __forceinline__ void atomicMaxF(float* addr, float val) {
  if (val >= 0.f)
    atomicMax((int*)addr, __float_as_int(val));
  else
    atomicMin((unsigned int*)addr, __float_as_uint(val));
}

// Pass 1: alpha = leaky_relu(e_src[s]+e_dst[d]); stash alpha; segment max
__global__ void edge_max(const int* __restrict__ s32, const int* __restrict__ d32,
                         int Et, const float* __restrict__ e_src,
                         const float* __restrict__ e_dst,
                         float* __restrict__ aw, float* __restrict__ amax) {
  int e = blockIdx.x * blockDim.x + threadIdx.x;
  if (e >= Et) return;
  int d = d32[e];
  float t = e_src[s32[e]] + e_dst[d];
  float alpha = t > 0.f ? t : NEG_SLOPE * t;
  aw[e] = alpha;
  atomicMaxF(&amax[d], alpha);
}

// Pass 2: ex = exp(alpha - amax[d]); stash ex; segment sum
__global__ void edge_sum(const int* __restrict__ d32, int Et,
                         const float* __restrict__ amax,
                         float* __restrict__ aw, float* __restrict__ denom) {
  int e = blockIdx.x * blockDim.x + threadIdx.x;
  if (e >= Et) return;
  int d = d32[e];
  float ex = expf(aw[e] - amax[d]);
  aw[e] = ex;
  atomicAdd(&denom[d], ex);
}

// Pass 3: wave per edge, lane per feature: accum[d,:] += (ex/denom[d]) * h[s,:]
__global__ void edge_scatter(const int* __restrict__ s32,
                             const int* __restrict__ d32, int Et,
                             const float* __restrict__ aw,
                             const float* __restrict__ denom,
                             const float* __restrict__ H,
                             float* __restrict__ accum) {
  long gid = (long)blockIdx.x * blockDim.x + threadIdx.x;
  int e    = (int)(gid >> 5);
  int lane = (int)(gid & 31);
  if (e >= Et) return;
  int s = s32[e];
  int d = d32[e];
  float coef = aw[e] / denom[d];            // wave-uniform broadcast loads
  float v = H[(size_t)s * 32 + lane] * coef;
  atomicAdd(&accum[(size_t)d * 32 + lane], v);
}

// out[i,j] = relu(accum[i,j] + b[j])  (float4 over n*8 vectors)
__global__ void bias_relu(const float* __restrict__ accum,
                          const float* __restrict__ b,
                          float* __restrict__ out, int n) {
  int idx = blockIdx.x * blockDim.x + threadIdx.x;
  if (idx >= n * 8) return;
  float4 v = ((const float4*)accum)[idx];
  float4 bb = ((const float4*)b)[idx & 7];
  v.x = fmaxf(v.x + bb.x, 0.f);
  v.y = fmaxf(v.y + bb.y, 0.f);
  v.z = fmaxf(v.z + bb.z, 0.f);
  v.w = fmaxf(v.w + bb.w, 0.f);
  ((float4*)out)[idx] = v;
}

// out[i,j] = maybe_sigmoid(lin[i,j] + lb[j])
__global__ void final_out(const float* __restrict__ lin,
                          const float* __restrict__ lb,
                          float* __restrict__ out, int n, int do_sigmoid) {
  int idx = blockIdx.x * blockDim.x + threadIdx.x;
  if (idx >= n * 8) return;
  float4 v = ((const float4*)lin)[idx];
  float4 bb = ((const float4*)lb)[idx & 7];
  v.x += bb.x; v.y += bb.y; v.z += bb.z; v.w += bb.w;
  if (do_sigmoid) {
    v.x = 1.f / (1.f + expf(-v.x));
    v.y = 1.f / (1.f + expf(-v.y));
    v.z = 1.f / (1.f + expf(-v.z));
    v.w = 1.f / (1.f + expf(-v.w));
  }
  ((float4*)out)[idx] = v;
}

// ---------------------------------------------------------------------------
// Host-side orchestration
// ---------------------------------------------------------------------------
static void run_conv(const float* xin, int K, const float* W, const float* as_,
                     const float* ad_, const int* s32, const int* d32,
                     int Et, int n, float* H, float* accum, float* aw,
                     float* e_s, float* e_d, float* amx, float* den,
                     hipStream_t stream) {
  dim3 blk(256);
  int waves = (n + 15) / 16;
  gat_gemm_wmma<<<dim3((waves + 7) / 8), blk, 0, stream>>>(xin, W, H, n, K);
  att_init<<<dim3((n + 255) / 256), blk, 0, stream>>>(H, as_, ad_, e_s, e_d,
                                                      amx, den, accum, n);
  dim3 grdE((Et + 255) / 256);
  edge_max<<<grdE, blk, 0, stream>>>(s32, d32, Et, e_s, e_d, aw, amx);
  edge_sum<<<grdE, blk, 0, stream>>>(d32, Et, amx, aw, den);
  long tot = (long)Et * 32;
  edge_scatter<<<dim3((unsigned)((tot + 255) / 256)), blk, 0, stream>>>(
      s32, d32, Et, aw, den, H, accum);
}

extern "C" void kernel_launch(void* const* d_in, const int* in_sizes, int n_in,
                              void* d_out, int out_size, void* d_ws, size_t ws_size,
                              hipStream_t stream) {
  (void)n_in; (void)out_size; (void)ws_size;

  const float*     x  = (const float*)d_in[0];
  const long long* ei = (const long long*)d_in[1];
  int n = in_sizes[0] / 64;     // IN = 64
  int E = in_sizes[1] / 2;      // edge_index is [2, E]
  int Et = E + n;               // with self-loops

  // workspace layout (f32 words unless noted):
  //   H | accum | xact | e_src | e_dst | amax | denom | s32 | d32 | aw
  float* ws    = (float*)d_ws;
  size_t n32   = (size_t)n * 32;
  float* H     = ws;
  float* accum = ws + n32;
  float* xact  = ws + 2 * n32;
  float* e_s   = ws + 3 * n32;
  float* e_d   = e_s + n;
  float* amx   = e_d + n;
  float* den   = amx + n;
  int*   s32   = (int*)(den + n);
  int*   d32   = s32 + Et;
  float* aw    = (float*)(d32 + Et);

  float* out = (float*)d_out;
  dim3 blk(256);
  dim3 grdV4((unsigned)((n * 8 + 255) / 256));

  // one-time edge-list conversion (int64 -> int32, self-loops appended)
  edge_prep<<<dim3((Et + 255) / 256), blk, 0, stream>>>(ei, ei + E, E, n,
                                                        s32, d32);

  for (int br = 0; br < 2; ++br) {
    int base = 2 + br * 10;   // c_* at 2..11, r_* at 12..21
    const float* W1  = (const float*)d_in[base + 0];
    const float* as1 = (const float*)d_in[base + 1];
    const float* ad1 = (const float*)d_in[base + 2];
    const float* b1  = (const float*)d_in[base + 3];
    const float* W2  = (const float*)d_in[base + 4];
    const float* as2 = (const float*)d_in[base + 5];
    const float* ad2 = (const float*)d_in[base + 6];
    const float* b2  = (const float*)d_in[base + 7];
    const float* lW  = (const float*)d_in[base + 8];
    const float* lb  = (const float*)d_in[base + 9];
    float* obr = out + (size_t)br * n32;   // prob_zero then res
    int sigmoid = (br == 0) ? 1 : 0;

    // Layer 1: x [n x 64] -> accum; activate into xact
    run_conv(x, 64, W1, as1, ad1, s32, d32, Et, n,
             H, accum, aw, e_s, e_d, amx, den, stream);
    bias_relu<<<grdV4, blk, 0, stream>>>(accum, b1, xact, n);

    // Layer 2: xact [n x 32] -> accum; activate into xact (H free after scatter)
    run_conv(xact, 32, W2, as2, ad2, s32, d32, Et, n,
             H, accum, aw, e_s, e_d, amx, den, stream);
    bias_relu<<<grdV4, blk, 0, stream>>>(accum, b2, xact, n);

    // Linear head: H = xact @ lW ; out = act(H + lb)
    int waves = (n + 15) / 16;
    gat_gemm_wmma<<<dim3((waves + 7) / 8), blk, 0, stream>>>(xact, lW, H, n, 32);
    final_out<<<grdV4, blk, 0, stream>>>(H, lb, obr, n, sigmoid);
  }
}
